// ScaledDotProductAttention_29738353558234
// MI455X (gfx1250) — matrix operand order repaired and checked by reading, not compiled, not run on hardware
//
#include <hip/hip_runtime.h>

typedef _Float16 v16h __attribute__((ext_vector_type(16)));
typedef float    v8f  __attribute__((ext_vector_type(8)));

// D = A(16x32 f16) * B(32x16 f16) + C(16x16 f32)
#define WMMA_F16(a, b, c) \
  __builtin_amdgcn_wmma_f32_16x16x32_f16(false, (a), false, (b), (short)0, (c), false, false)

constexpr int   S_LEN  = 2048;
constexpr int   D_DIM  = 64;
constexpr int   BH     = 32;     // B*H
constexpr int   QTILES = 64;     // S / 32 query blocks per head
// 1/sqrt(64) * log2(e): scores land in the exp2 domain (softmax is invariant)
constexpr float SCALE_LOG2E = 0.125f * 1.44269504088896340736f;
constexpr float NEGINF      = -1.0e9f;

// ---- 16-lane max reduction on the VALU via DPP row_ror (row = 16 lanes) ----
template <int CTRL>
__device__ __forceinline__ float dppmov(float x) {
  return __builtin_bit_cast(float,
      __builtin_amdgcn_update_dpp(0, __builtin_bit_cast(int, x), CTRL, 0xf, 0xf, true));
}
__device__ __forceinline__ float rmax16(float x) {
  x = fmaxf(x, dppmov<0x128>(x));  // row_ror:8
  x = fmaxf(x, dppmov<0x124>(x));  // row_ror:4
  x = fmaxf(x, dppmov<0x122>(x));  // row_ror:2
  x = fmaxf(x, dppmov<0x121>(x));  // row_ror:1
  return x;
}

// ---------------------------------------------------------------------------
// Pass 1: LDS-tiled transpose + f32->f16 convert.
//   in[bh][r][c] (f32, rows x cols)  ->  out[bh][c][r] (f16, cols x rows)
// ---------------------------------------------------------------------------
__launch_bounds__(256)
__global__ void transpose_cvt_f16(const float* __restrict__ in,
                                  _Float16* __restrict__ out,
                                  int rows, int cols) {
  __shared__ float tile[32][33];
  const int tilesC     = cols >> 5;
  const int tilesPlane = (rows >> 5) * tilesC;
  const int bh = blockIdx.x / tilesPlane;
  const int tt = blockIdx.x % tilesPlane;
  const int r0 = (tt / tilesC) << 5;
  const int c0 = (tt % tilesC) << 5;

  const float* ip = in  + (size_t)bh * rows * cols;
  _Float16*    op = out + (size_t)bh * rows * cols;
  const int x = threadIdx.x;   // 0..31
  const int y = threadIdx.y;   // 0..7

#pragma unroll
  for (int yy = y; yy < 32; yy += 8)
    tile[yy][x] = ip[(size_t)(r0 + yy) * cols + (c0 + x)];
  __syncthreads();
#pragma unroll
  for (int yy = y; yy < 32; yy += 8)
    op[(size_t)(c0 + yy) * rows + (r0 + x)] = (_Float16)tile[x][yy];
}

// ---------------------------------------------------------------------------
// Pass 2: causal flash attention, WMMA f16 / f32 accumulate.
//   Per wave: 32 query rows, KV tiles of 64 columns (only last tile masked).
//   Softmax denominator is accumulated in the MATRIX pipe: an extra
//   accumulator tile per m-tile gets P @ ones, so no cross-lane sum
//   reduction is ever needed (row max still uses DPP).
// ---------------------------------------------------------------------------
__launch_bounds__(256)
__global__ void flash_attn_causal_wmma(const float* __restrict__ q,
                                       const _Float16* __restrict__ kt,
                                       const _Float16* __restrict__ vt,
                                       float* __restrict__ out) {
  // Per-wave P staging: 8 waves * 2 m-tiles * (16x64 f16) = 32 KB
  __shared__ __align__(64) _Float16 plds[8][2][16 * 64];

  const int lane = threadIdx.x & 31;
  const int wib  = threadIdx.x >> 5;                 // wave in block
  const int wid  = blockIdx.x * 8 + wib;             // global wave id
  const int bh   = wid >> 6;                         // head index (0..31)
  const int t    = wid & 63;                         // 32-row query block (0..63)
  if (bh >= BH) return;

  const int half = lane >> 4;                        // 16-lane group id
  const int n    = lane & 15;                        // lane within group
  const int r0   = t * 32;                           // first query row of this wave

  const float*    qb = q   + (size_t)bh * S_LEN * D_DIM;
  const _Float16* kb = kt  + (size_t)bh * S_LEN * D_DIM;  // (S, 64) f16
  const _Float16* vb = vt  + (size_t)bh * D_DIM * S_LEN;  // (64, S) f16
  float*          ob = out + (size_t)bh * S_LEN * D_DIM;

  // ---- Resident Q tiles in WMMA A-layout, pre-scaled, f32 -> f16 ----
  v16h aQ[2][2];
#pragma unroll
  for (int mt = 0; mt < 2; ++mt) {
    const float* qrow = qb + (size_t)(r0 + mt * 16 + n) * D_DIM;
#pragma unroll
    for (int kc = 0; kc < 2; ++kc) {
      const int b0 = kc * 32 + 16 * half;
      const int b1 = kc * 32 + 16 * half + 8;
#pragma unroll
      for (int i = 0; i < 8; ++i) {
        aQ[mt][kc][i]     = (_Float16)(qrow[b0 + i] * SCALE_LOG2E);
        aQ[mt][kc][i + 8] = (_Float16)(qrow[b1 + i] * SCALE_LOG2E);
      }
    }
  }

  // All-ones B matrix for the denominator accumulation (P @ 1 = row sums)
  v16h bOnes;
#pragma unroll
  for (int i = 0; i < 16; ++i) bOnes[i] = (_Float16)1.0f;

  // ---- Online-softmax state (exp2 domain) ----
  float mrun[2][8];
  v8f   Oacc[2][4];   // output numerator tiles
  v8f   Lacc[2];      // denominator tiles (every column = row sum)
#pragma unroll
  for (int mt = 0; mt < 2; ++mt) {
#pragma unroll
    for (int vv = 0; vv < 8; ++vv) mrun[mt][vv] = -3.0e38f;
#pragma unroll
    for (int w = 0; w < 4; ++w) { v8f z = {}; Oacc[mt][w] = z; }
    { v8f z = {}; Lacc[mt] = z; }
  }

  _Float16* slab[2] = { &plds[wib][0][0], &plds[wib][1][0] };
  // P-store offset within a 32-col chunk (natural column order):
  //   off(g, n) = (g>>1)*32 + (g & 1)*16 + n
  const int pbase = n;
  const int jmax  = t >> 1;   // last 64-wide KV tile index (diagonal)

  for (int j = 0; j <= jmax; ++j) {
    const int col0 = j * 64;

    if (j < jmax) {  // hint next KV tile toward the caches
      __builtin_prefetch(kb + (size_t)(col0 + 64 + n) * D_DIM, 0, 0);
      __builtin_prefetch(vb + (size_t)n * S_LEN + col0 + 64, 0, 0);
    }

    // ---- Scores (log2-scaled): S = (Q*scale*log2e) @ K^T ----
    v8f s[2][4];
#pragma unroll
    for (int g = 0; g < 4; ++g) {
      const _Float16* kp = kb + (size_t)(col0 + g * 16 + n) * D_DIM + 16 * half;
      const v16h bk0 = *(const v16h*)(kp);        // d = 16*half + 0..15
      const v16h bk1 = *(const v16h*)(kp + 32);   // d = 32 + 16*half + 0..15
#pragma unroll
      for (int mt = 0; mt < 2; ++mt) {
        v8f acc = {};
        acc = WMMA_F16(aQ[mt][0], bk0, acc);
        acc = WMMA_F16(aQ[mt][1], bk1, acc);
        s[mt][g] = acc;
      }
    }

    // ---- Causal mask on the diagonal tile ----
    if (j == jmax) {
#pragma unroll
      for (int mt = 0; mt < 2; ++mt)
#pragma unroll
        for (int g = 0; g < 4; ++g)
#pragma unroll
          for (int vv = 0; vv < 8; ++vv)
            if (col0 + g * 16 + n > r0 + mt * 16 + vv + 8 * half) s[mt][g][vv] = NEGINF;
    }

    // ---- Online softmax (exp2 domain); write P (f16) to row-major LDS slab ----
#pragma unroll
    for (int mt = 0; mt < 2; ++mt) {
#pragma unroll
      for (int vv = 0; vv < 8; ++vv) {
        float mx = fmaxf(fmaxf(s[mt][0][vv], s[mt][1][vv]),
                         fmaxf(s[mt][2][vv], s[mt][3][vv]));
        mx = rmax16(mx);
        const float newm  = fmaxf(mrun[mt][vv], mx);
        const float alpha = __builtin_amdgcn_exp2f(mrun[mt][vv] - newm);
        mrun[mt][vv] = newm;
        float p[4];
#pragma unroll
        for (int g = 0; g < 4; ++g)
          p[g] = __builtin_amdgcn_exp2f(s[mt][g][vv] - newm);
        // Rescale numerator AND denominator accumulators by alpha.
#pragma unroll
        for (int w = 0; w < 4; ++w) Oacc[mt][w][vv] *= alpha;
        Lacc[mt][vv] *= alpha;
        _Float16* sp = slab[mt] + (vv + 8 * half) * 64 + pbase;
#pragma unroll
        for (int g = 0; g < 4; ++g)
          sp[(g >> 1) * 32 + (g & 1) * 16] = (_Float16)p[g];
      }
    }

    asm volatile("s_wait_dscnt 0" ::: "memory");  // intra-wave LDS store->load ordering

    // ---- Reload P in A-layout: one contiguous 32B LDS load per (mt, chunk) ----
    v16h aP[2][2];
#pragma unroll
    for (int mt = 0; mt < 2; ++mt) {
      const _Float16* sp = slab[mt] + n * 64 + 16 * half;  // row M = n
      aP[mt][0] = *(const v16h*)(sp);
      aP[mt][1] = *(const v16h*)(sp + 32);
    }
    asm volatile("" ::: "memory");  // keep next iter's LDS stores after these loads

    // ---- Denominator: Lacc += P @ ones (matrix pipe does the row sums) ----
#pragma unroll
    for (int mt = 0; mt < 2; ++mt) {
      Lacc[mt] = WMMA_F16(aP[mt][0], bOnes, Lacc[mt]);
      Lacc[mt] = WMMA_F16(aP[mt][1], bOnes, Lacc[mt]);
    }

    // ---- O += P @ V : load each V tile pair and consume immediately ----
#pragma unroll
    for (int w = 0; w < 4; ++w) {
      const _Float16* vp = vb + (size_t)(w * 16 + n) * S_LEN + col0 + 16 * half;
      const v16h bv0 = *(const v16h*)(vp);        // kv = col0 + 16*half + 0..15
      const v16h bv1 = *(const v16h*)(vp + 32);   // kv = col0 + 32 + 16*half + 0..15
#pragma unroll
      for (int mt = 0; mt < 2; ++mt) {
        Oacc[mt][w] = WMMA_F16(aP[mt][0], bv0, Oacc[mt][w]);
        Oacc[mt][w] = WMMA_F16(aP[mt][1], bv1, Oacc[mt][w]);
      }
    }
  }

  // ---- Normalize and store (every Lacc column already holds the row sum) ----
#pragma unroll
  for (int mt = 0; mt < 2; ++mt) {
#pragma unroll
    for (int vv = 0; vv < 8; ++vv) {
      const float inv = 1.0f / Lacc[mt][vv];
      const int   row = r0 + mt * 16 + vv + 8 * half;
      float* op = ob + (size_t)row * D_DIM;
#pragma unroll
      for (int w = 0; w < 4; ++w)
        op[w * 16 + n] = Oacc[mt][w][vv] * inv;
    }
  }
}

extern "C" void kernel_launch(void* const* d_in, const int* in_sizes, int n_in,
                              void* d_out, int out_size, void* d_ws, size_t ws_size,
                              hipStream_t stream) {
  (void)in_sizes; (void)n_in; (void)out_size; (void)ws_size;
  const float* q = (const float*)d_in[0];
  const float* k = (const float*)d_in[1];  // (B,H,D,S) pre-transposed
  const float* v = (const float*)d_in[2];
  float* out = (float*)d_out;

  // Workspace: kT (S,64) f16 then vT (64,S) f16 per head = 16 MB total.
  _Float16* kt = (_Float16*)d_ws;
  _Float16* vt = kt + (size_t)BH * S_LEN * D_DIM;

  // Pass 1: convert+transpose K and V to f16 WMMA-friendly layouts.
  {
    dim3 blk(32, 8);
    dim3 gK(BH * (D_DIM / 32) * (S_LEN / 32));  // K: (64,2048) -> (2048,64)
    hipLaunchKernelGGL(transpose_cvt_f16, gK, blk, 0, stream, k, kt, D_DIM, S_LEN);
    dim3 gV(BH * (S_LEN / 32) * (D_DIM / 32));  // V: (2048,64) -> (64,2048)
    hipLaunchKernelGGL(transpose_cvt_f16, gV, blk, 0, stream, v, vt, S_LEN, D_DIM);
  }

  // Pass 2: 32 heads * 64 query-blocks = 2048 waves; 8 waves (256 threads) per block
  dim3 grid(BH * QTILES / 8), block(256);
  hipLaunchKernelGGL(flash_attn_causal_wmma, grid, block, 0, stream, q, kt, vt, out);
}
